// MultiHeadAttention_43267500540681
// MI455X (gfx1250) — matrix operand order repaired and checked by reading, not compiled, not run on hardware
//
#include <hip/hip_runtime.h>
#include <hip/hip_bf16.h>

// ---------- types ----------
typedef _Float16 half4  __attribute__((ext_vector_type(4)));
typedef _Float16 half8  __attribute__((ext_vector_type(8)));
typedef _Float16 half16 __attribute__((ext_vector_type(16)));
typedef float    v8f    __attribute__((ext_vector_type(8)));
typedef float    f32x4  __attribute__((ext_vector_type(4)));

union H16 { half16 v; half8 h[2]; };

#define WMMA(a, b, c) \
  __builtin_amdgcn_wmma_f32_16x16x32_f16(false, (a), false, (b), (short)0, (c), false, false)

// Problem constants
#define BB 4
#define SS 1024
#define DD 1024
#define HH 16
#define DH 64
#define MM (BB * SS)   // 4096 flattened (b,s) rows

// ---------- kernel 0: fp32 -> fp16 elementwise (vectorized x4) ----------
__global__ __launch_bounds__(256) void cvt_f32_to_f16(const float* __restrict__ in,
                                                      _Float16* __restrict__ out) {
  int i = blockIdx.x * 256 + threadIdx.x;
  f32x4 f = *(const f32x4*)(in + (size_t)i * 4);
  half4 h;
  h[0] = (_Float16)f[0]; h[1] = (_Float16)f[1];
  h[2] = (_Float16)f[2]; h[3] = (_Float16)f[3];
  *(half4*)(out + (size_t)i * 4) = h;
}

// ---------- kernel 1: weight transpose fp32 -> fp16, Wt[n][k] = W[k][n] ----------
__global__ __launch_bounds__(256) void transpose_w_f16(const float* __restrict__ W,
                                                       _Float16* __restrict__ Wt) {
  int idx = blockIdx.x * 256 + threadIdx.x;  // idx = n*1024 + k  (writes coalesced)
  int n = idx >> 10;
  int k = idx & 1023;
  Wt[idx] = (_Float16)W[k * DD + n];
}

// ---------- kernel 2: WMMA GEMM  C[4096x1024] = X[4096x1024] * W ----------
// Wave tile: 32 rows x 64 cols, K pipelined in 32-chunks with double-buffered
// fragments (8 back-to-back WMMAs per k-step; next-step loads overlap them).
// X: fp16 row-major. Wt: fp16 transposed (Wt[n][k]).
// mode 0: outH = f16 [B,H,S,dh]      (Q, K projections, head-split)
// mode 1: outH = f16 [B,H,dh,S]      (V projection, transposed per head)
// mode 2: outF = f32 row-major [4096,1024]  (final output projection)
__global__ __launch_bounds__(256) void gemm_wmma(const _Float16* __restrict__ X,
                                                 const _Float16* __restrict__ Wt,
                                                 _Float16* __restrict__ outH,
                                                 float* __restrict__ outF, int mode) {
  int wid  = blockIdx.x * 8 + (threadIdx.x >> 5);  // 2048 waves total
  int lane = threadIdx.x & 31;
  int m    = lane & 15;
  int hi   = lane >> 4;             // 0 / 1 half-wave
  int rowBase = (wid >> 4) * 32;    // 128 row tiles of 32
  int colBase = (wid & 15) * 64;    // 16 col tiles of 64
  int asel = hi ? 8 : 0;            // A-frag K-subset select
  int bsel = hi ? 8 : 0;            // B-frag K-subset select

  const _Float16* xr0 = X + (size_t)(rowBase + m) * DD + asel;
  const _Float16* xr1 = X + (size_t)(rowBase + 16 + m) * DD + asel;
  const _Float16* wr  = Wt + (size_t)(colBase + m) * DD + bsel;  // + g*16*DD per group

  v8f acc0[4] = {};
  v8f acc1[4] = {};

  H16 Aa0, Aa1, Ab0, Ab1;
  half16 Ba[4], Bb[4];

#define LOADK(A0, A1, Bv, K0)                                            \
  do {                                                                   \
    A0.h[0] = *(const half8*)(xr0 + (K0));                               \
    A0.h[1] = *(const half8*)(xr0 + (K0) + 16);                          \
    A1.h[0] = *(const half8*)(xr1 + (K0));                               \
    A1.h[1] = *(const half8*)(xr1 + (K0) + 16);                          \
    _Pragma("unroll") for (int g = 0; g < 4; ++g)                        \
        Bv[g] = __builtin_shufflevector(*(const half8*)(wr + (size_t)g * 16 * DD + (K0)), *(const half8*)(wr + (size_t)g * 16 * DD + (K0) + 16), 0,1,2,3,4,5,6,7,8,9,10,11,12,13,14,15); \
  } while (0)

#define COMPK(A0, A1, Bv)                                                \
  do {                                                                   \
    _Pragma("unroll") for (int g = 0; g < 4; ++g)                        \
        acc0[g] = WMMA(A0.v, Bv[g], acc0[g]);                            \
    _Pragma("unroll") for (int g = 0; g < 4; ++g)                        \
        acc1[g] = WMMA(A1.v, Bv[g], acc1[g]);                            \
  } while (0)

  LOADK(Aa0, Aa1, Ba, 0);
  for (int k0 = 32; k0 < DD - 32; k0 += 64) {
    LOADK(Ab0, Ab1, Bb, k0);        // prefetch slot b while computing slot a
    COMPK(Aa0, Aa1, Ba);
    LOADK(Aa0, Aa1, Ba, k0 + 32);   // prefetch slot a while computing slot b
    COMPK(Ab0, Ab1, Bb);
  }
  LOADK(Ab0, Ab1, Bb, DD - 32);
  COMPK(Aa0, Aa1, Ba);
  COMPK(Ab0, Ab1, Bb);
#undef LOADK
#undef COMPK

  auto store_one = [&](float v, int r, int c) {
    if (mode == 2) {
      outF[(size_t)r * DD + c] = v;
    } else {
      int b = r >> 10, s = r & 1023;     // r = b*S + s
      int hd = c >> 6, d = c & 63;       // c = hd*dh + d
      if (mode == 0)
        outH[(((size_t)(b * HH + hd) * SS) + s) * DH + d] = (_Float16)v;
      else
        outH[(((size_t)(b * HH + hd) * DH) + d) * SS + s] = (_Float16)v;
    }
  };

#pragma unroll
  for (int g = 0; g < 4; ++g) {
#pragma unroll
    for (int i = 0; i < 8; ++i) {
      int c = colBase + g * 16 + m;
      store_one(acc0[g][i], rowBase + i + hi * 8, c);
      store_one(acc1[g][i], rowBase + 16 + i + hi * 8, c);
    }
  }
}

// ---------- kernel 3: attention (one wave per (b, h, 16-query tile)) ----------
// Qh, Kh: f16 [B,H,S,dh]; Vt: f16 [B,H,dh,S].
// Scores computed transposed: S^T = K_tile(16k x 64d) * Q^T(64d x 16q)
// => C-layout of S^T (lane = query col, vreg = key row) is exactly the
//    A-fragment layout needed for P x V. No transpose needed.
__global__ __launch_bounds__(256) void attn_wmma(const _Float16* __restrict__ Qh,
                                                 const _Float16* __restrict__ Kh,
                                                 const _Float16* __restrict__ Vt,
                                                 const int* __restrict__ valid_lens,
                                                 _Float16* __restrict__ concat) {
  int wid  = blockIdx.x * 8 + (threadIdx.x >> 5);  // 0..4095
  int lane = threadIdx.x & 31;
  int m    = lane & 15;
  int hi   = lane >> 4;
  int b  = wid >> 10;
  int h  = (wid >> 6) & 15;
  int qBase = (wid & 63) * 16;
  int valid = valid_lens[b];

  const _Float16* Qb = Qh + (size_t)(b * HH + h) * SS * DH;
  const _Float16* Kb = Kh + (size_t)(b * HH + h) * SS * DH;
  const _Float16* Vb = Vt + (size_t)(b * HH + h) * DH * SS;

  // Q as B operand: lane holds column q=m, d = 8h+0..7 and 16+8h+0..7 per fragment.
  half16 bq0, bq1;
  {
    const _Float16* qrow = Qb + (size_t)(qBase + m) * DH;
    int ks = hi ? 8 : 0;
    bq0 = __builtin_shufflevector(*(const half8*)(qrow + ks), *(const half8*)(qrow + ks + 16), 0,1,2,3,4,5,6,7,8,9,10,11,12,13,14,15);        // d 0..31 chunk
    bq1 = __builtin_shufflevector(*(const half8*)(qrow + 32 + ks), *(const half8*)(qrow + 48 + ks), 0,1,2,3,4,5,6,7,8,9,10,11,12,13,14,15);   // d 32..63 chunk
  }

  v8f accO[4] = {};
  float lsum = 0.f;
  int asel = hi ? 8 : 0;
  int bsel = hi ? 8 : 0;
  int kend = ((valid + 31) >> 5) << 5;  // masked tail tiles contribute exactly 0
  if (kend > SS) kend = SS;

  for (int kb = 0; kb < kend; kb += 32) {
    const _Float16* krow0 = Kb + (size_t)(kb + m) * DH + asel;       // keys kb..kb+15
    const _Float16* krow1 = Kb + (size_t)(kb + 16 + m) * DH + asel;  // keys kb+16..kb+31
    const _Float16* vrow  = Vb + (size_t)m * SS + kb + bsel;

    // ---- K A-fragments (row = key) for both 16-key subtiles, both d-chunks
    H16 a00, a01, a10, a11;
    a00.h[0] = *(const half8*)(krow0);
    a00.h[1] = *(const half8*)(krow0 + 16);
    a01.h[0] = *(const half8*)(krow0 + 32);
    a01.h[1] = *(const half8*)(krow0 + 48);
    a10.h[0] = *(const half8*)(krow1);
    a10.h[1] = *(const half8*)(krow1 + 16);
    a11.h[0] = *(const half8*)(krow1 + 32);
    a11.h[1] = *(const half8*)(krow1 + 48);

    // ---- V B-fragments hoisted: independent of scores, overlap score WMMAs/exp
    half16 bv[4];
#pragma unroll
    for (int g = 0; g < 4; ++g)
      bv[g] = __builtin_shufflevector(*(const half8*)(vrow + (size_t)g * 16 * SS), *(const half8*)(vrow + (size_t)g * 16 * SS + 16), 0,1,2,3,4,5,6,7,8,9,10,11,12,13,14,15);

    // ---- prefetch next key tile (global_prefetch_b8; pulls lines toward WGP)
    __builtin_prefetch(krow0 + 32 * DH, 0, 1);
    __builtin_prefetch(krow1 + 32 * DH, 0, 1);
    __builtin_prefetch(vrow + 32, 0, 1);

    // ---- scores transposed: two 16x16 tiles
    v8f sc0, sc1;
    {
      v8f z = {};
      z   = WMMA(a00.v, bq0, z);
      sc0 = WMMA(a01.v, bq1, z);
    }
    {
      v8f z = {};
      z   = WMMA(a10.v, bq0, z);
      sc1 = WMMA(a11.v, bq1, z);
    }

    // ---- softmax numerator + mask, packed directly as P's A-fragment
    H16 p;
#pragma unroll
    for (int i = 0; i < 8; ++i) {
      int key0 = kb + i + asel;
      float e0 = (key0 < valid) ? __expf(sc0[i] * 0.125f) : 0.f;  // 1/sqrt(64)
      lsum += e0;
      p.v[i] = (_Float16)e0;
      int key1 = kb + 16 + i + asel;
      float e1 = (key1 < valid) ? __expf(sc1[i] * 0.125f) : 0.f;
      lsum += e1;
      p.v[8 + i] = (_Float16)e1;
    }

    // ---- O += P x V
#pragma unroll
    for (int g = 0; g < 4; ++g)
      accO[g] = WMMA(p.v, bv[g], accO[g]);
  }

  // Each lane summed half the keys for its query column; combine halves.
  lsum += __shfl_xor(lsum, 16, 32);

  // accO rows are queries i + hi*8; l for query r lives in lane r (and r+16).
  float linv[8];
#pragma unroll
  for (int i = 0; i < 8; ++i) {
    float l = __shfl(lsum, i + hi * 8, 32);
    linv[i] = (l > 0.f) ? (1.f / l) : 0.f;
  }

  // Head-interleaved concat: concat[b, s, d*H + h]  (reference permute(0,2,3,1))
#pragma unroll
  for (int g = 0; g < 4; ++g) {
#pragma unroll
    for (int i = 0; i < 8; ++i) {
      int s  = qBase + i + hi * 8;
      int dl = g * 16 + m;
      concat[((size_t)(b * SS + s)) * DD + dl * HH + h] =
          (_Float16)(accO[g][i] * linv[i]);
    }
  }
}

// ---------- host launcher ----------
extern "C" void kernel_launch(void* const* d_in, const int* in_sizes, int n_in,
                              void* d_out, int out_size, void* d_ws, size_t ws_size,
                              hipStream_t stream) {
  const float* q  = (const float*)d_in[0];
  const float* k  = (const float*)d_in[1];
  const float* v  = (const float*)d_in[2];
  const int*   vl = (const int*)d_in[3];
  const float* Wq = (const float*)d_in[4];
  const float* Wk = (const float*)d_in[5];
  const float* Wv = (const float*)d_in[6];
  const float* Wo = (const float*)d_in[7];

  char* ws = (char*)d_ws;
  const size_t MB = 1ull << 20;
  _Float16* qx     = (_Float16*)(ws + 0 * MB);   // [4096,1024] f16
  _Float16* kx     = (_Float16*)(ws + 8 * MB);
  _Float16* vx     = (_Float16*)(ws + 16 * MB);
  _Float16* wqT    = (_Float16*)(ws + 24 * MB);  // [n][k] f16, 2MB each
  _Float16* wkT    = (_Float16*)(ws + 26 * MB);
  _Float16* wvT    = (_Float16*)(ws + 28 * MB);
  _Float16* woT    = (_Float16*)(ws + 30 * MB);
  _Float16* Qhead  = (_Float16*)(ws + 32 * MB);  // [B,H,S,dh] f16
  _Float16* Khead  = (_Float16*)(ws + 40 * MB);  // [B,H,S,dh] f16
  _Float16* Vtr    = (_Float16*)(ws + 48 * MB);  // [B,H,dh,S] f16
  _Float16* concat = (_Float16*)(ws + 56 * MB);  // [4096,1024] f16, head-interleaved

  // activations -> f16 (4M elems each, x4 vectorized)
  cvt_f32_to_f16<<<4096, 256, 0, stream>>>(q, qx);
  cvt_f32_to_f16<<<4096, 256, 0, stream>>>(k, kx);
  cvt_f32_to_f16<<<4096, 256, 0, stream>>>(v, vx);

  // weights -> f16 transposed (1M elems each)
  transpose_w_f16<<<4096, 256, 0, stream>>>(Wq, wqT);
  transpose_w_f16<<<4096, 256, 0, stream>>>(Wk, wkT);
  transpose_w_f16<<<4096, 256, 0, stream>>>(Wv, wvT);
  transpose_w_f16<<<4096, 256, 0, stream>>>(Wo, woT);

  // projections (WMMA): Q,K head-split; V head-split + transposed
  gemm_wmma<<<256, 256, 0, stream>>>(qx, wqT, Qhead, nullptr, 0);
  gemm_wmma<<<256, 256, 0, stream>>>(kx, wkT, Khead, nullptr, 0);
  gemm_wmma<<<256, 256, 0, stream>>>(vx, wvT, Vtr,   nullptr, 1);

  // fused masked-softmax attention (WMMA)
  attn_wmma<<<512, 256, 0, stream>>>(Qhead, Khead, Vtr, vl, concat);

  // output projection -> fp32 d_out
  gemm_wmma<<<256, 256, 0, stream>>>(concat, woT, nullptr, (float*)d_out, 2);
}
